// BahdanauAttention_59227599012441
// MI455X (gfx1250) — matrix operand order repaired and checked by reading, not compiled, not run on hardware
//
#include <hip/hip_runtime.h>

// ---------------------------------------------------------------------------
// B=32, L=2048, D=1024, E=2048
//  in: dec_hidden[B,D] f32, enc_outputs[B,L,E] f32, W_s[D,D] f32,
//      W_h[D,E] f32, v[D] f32
//  out: ctx[B,E] f32 ++ attn[B,L] f32
// ---------------------------------------------------------------------------

#define Bn 32
#define Ln 2048
#define Dn 1024
#define En 2048

typedef __attribute__((ext_vector_type(16))) __bf16 v16bf;
typedef __attribute__((ext_vector_type(8)))  float  v8f;

union Frag { uint4 u[2]; v16bf v; };

#if __has_builtin(__builtin_amdgcn_tanhf)
#define TANHF(x) __builtin_amdgcn_tanhf(x)
#else
#define TANHF(x) tanhf(x)
#endif

static __device__ __forceinline__ unsigned short f2bf(float f) {
    unsigned int u = __float_as_uint(f);
    unsigned int r = u + 0x7FFFu + ((u >> 16) & 1u);   // round-to-nearest-even
    return (unsigned short)(r >> 16);
}

// --------------------------- kernel 0: W_h -> bf16 -------------------------
__global__ void wh_cvt_kernel(const float* __restrict__ Wh,
                              unsigned short* __restrict__ Whb) {
    int t = blockIdx.x * 256 + threadIdx.x;          // t < D*E/4 = 524288
    float4 f = ((const float4*)Wh)[t];
    ushort4 h;
    h.x = f2bf(f.x); h.y = f2bf(f.y); h.z = f2bf(f.z); h.w = f2bf(f.w);
    ((ushort4*)Whb)[t] = h;
}

// ------------------- kernel 1: s[b,e'] = dec[b,:] . W_s[e',:] --------------
__global__ void sproj_kernel(const float* __restrict__ dec,
                             const float* __restrict__ Ws,
                             float* __restrict__ s_ws) {
    int t = blockIdx.x * 256 + threadIdx.x;          // t < B*D = 32768
    int b = t >> 10, e = t & 1023;
    const float4* dv = (const float4*)(dec + (size_t)b * Dn);
    const float4* wv = (const float4*)(Ws + (size_t)e * Dn);
    float acc = 0.f;
    #pragma unroll 4
    for (int i = 0; i < Dn / 4; ++i) {
        float4 a = dv[i], w = wv[i];
        acc += a.x * w.x + a.y * w.y + a.z * w.z + a.w * w.w;
    }
    s_ws[t] = acc;
}

// --------- kernel 2: fused  scores[b,l] = v . tanh(s[b,:] + enc@W_h^T) -----
// grid (L/64, B), block 128 (4 waves). The 4 waves cooperatively stage a
// 64-row enc slab as bf16 in LDS (4 x 64KB). Then each wave computes ALL 64
// l-rows for ITS OWN 256 d-columns: per k-step, 4 A frags (shared LDS) x
// 4 B frags (W_h bf16 from L2) -> 16 wmmas, i.e. 1.0 loads/wmma and 4x
// B-reuse vs one-slab-per-wave. Scores finish with a tiny LDS reduction.
__global__ __launch_bounds__(128) void score_kernel(
        const float* __restrict__ enc,
        const unsigned short* __restrict__ whb,
        const float* __restrict__ s_ws,
        const float* __restrict__ vvec,
        float* __restrict__ scores) {
    extern __shared__ unsigned short smem[];   // 4*16*2048 bf16 + 256 floats
    const int tid  = threadIdx.x;
    const int w    = tid >> 5;
    const int lane = tid & 31;
    const int b    = blockIdx.y;
    const int lbase = blockIdx.x * 64;

    float* pred = (float*)(smem + 4 * 16 * En);      // [4 waves][64 rows]

    // ---- stage: wave w converts enc rows [lbase+w*16, +16) -> slab w ----
    {
        unsigned short* slab = smem + w * (16 * En);
        const float4* src =
            (const float4*)(enc + ((size_t)b * Ln + lbase + w * 16) * En);
        ushort4* dst = (ushort4*)slab;
        for (int i = lane; i < 16 * En / 4; i += 32) {
            float4 f = src[i];
            ushort4 h;
            h.x = f2bf(f.x); h.y = f2bf(f.y); h.z = f2bf(f.z); h.w = f2bf(f.w);
            dst[i] = h;
        }
    }
    __syncthreads();

    const uint4* slab4 = (const uint4*)smem;   // slab mt at offset mt*4096
    const uint4* wh4   = (const uint4*)whb;
    const int row = lane & 15;
    const int hi  = lane >> 4;
    const int dbase = w * 256;                 // this wave's d-column range

    float p[4][8];
    #pragma unroll
    for (int mt = 0; mt < 4; ++mt)
        #pragma unroll
        for (int j = 0; j < 8; ++j) p[mt][j] = 0.f;

    for (int dci = 0; dci < 4; ++dci) {
        const int d0 = dbase + dci * 64;
        v8f acc[4][4];                          // [m-tile][n-tile]
        #pragma unroll
        for (int mt = 0; mt < 4; ++mt)
            #pragma unroll
            for (int nt = 0; nt < 4; ++nt)
                acc[mt][nt] = (v8f){0.f,0.f,0.f,0.f,0.f,0.f,0.f,0.f};

        #pragma unroll 1
        for (int k0 = 0; k0 < En; k0 += 32) {
            // A frags: lane holds row (lane&15) of slab mt,
            //          K = {k0+hi*8 .. +7} and {+16 .. +23}
            Frag a[4];
            #pragma unroll
            for (int mt = 0; mt < 4; ++mt) {
                int c0 = mt * 4096 + row * 256 + (k0 >> 3) + hi;
                a[mt].u[0] = slab4[c0];
                a[mt].u[1] = slab4[c0 + 2];
            }
            #pragma unroll
            for (int nt = 0; nt < 4; ++nt) {
                // B frag: lane holds W_h row d0+nt*16+(lane&15),
                //         K = {k0+hi*8 .. +7} and {+16 .. +23}
                Frag bf;
                int drow = d0 + nt * 16 + row;
                size_t cb = (size_t)drow * 256 + (k0 >> 3) + hi;
                bf.u[0] = wh4[cb];
                bf.u[1] = wh4[cb + 2];
                #pragma unroll
                for (int mt = 0; mt < 4; ++mt)
                    acc[mt][nt] = __builtin_amdgcn_wmma_f32_16x16x32_bf16(
                        false, a[mt].v, false, bf.v, (short)0, acc[mt][nt],
                        false, false);
            }
        }

        // fold: p += v[c] * tanh(acc + s[b,c]) ; column c fixed per lane
        #pragma unroll
        for (int nt = 0; nt < 4; ++nt) {
            int c = d0 + nt * 16 + row;
            float sv = s_ws[b * Dn + c];
            float vv = vvec[c];
            #pragma unroll
            for (int mt = 0; mt < 4; ++mt)
                #pragma unroll
                for (int j = 0; j < 8; ++j)
                    p[mt][j] += vv * TANHF(acc[mt][nt][j] + sv);
        }
    }

    // ---- reduce 16 lanes of each half-wave; rows mt*16 + hi*8 + j ----
    #pragma unroll
    for (int mt = 0; mt < 4; ++mt) {
        #pragma unroll
        for (int j = 0; j < 8; ++j) {
            float x = p[mt][j];
            x += __shfl_xor(x, 1, 32);
            x += __shfl_xor(x, 2, 32);
            x += __shfl_xor(x, 4, 32);
            x += __shfl_xor(x, 8, 32);
            p[mt][j] = x;
        }
    }
    if (row == 0) {
        #pragma unroll
        for (int mt = 0; mt < 4; ++mt)
            #pragma unroll
            for (int j = 0; j < 8; ++j)
                pred[w * 64 + mt * 16 + hi * 8 + j] = p[mt][j];
    }
    __syncthreads();

    // ---- cross-wave d-partition sum -> final scores ----
    if (tid < 64) {
        float s = pred[tid] + pred[64 + tid] + pred[128 + tid] + pred[192 + tid];
        scores[b * Ln + lbase + tid] = s;
    }
}

// --------------------------- kernel 3: softmax -----------------------------
__global__ void softmax_kernel(const float* __restrict__ scores,
                               float* __restrict__ attn) {
    int b = blockIdx.x, tid = threadIdx.x;
    __shared__ float red[256];
    const float* sp = scores + b * Ln;
    float e[8];
    float m = -3.4e38f;
    #pragma unroll
    for (int i = 0; i < 8; ++i) { e[i] = sp[tid + i * 256]; m = fmaxf(m, e[i]); }
    red[tid] = m; __syncthreads();
    for (int s = 128; s > 0; s >>= 1) {
        if (tid < s) red[tid] = fmaxf(red[tid], red[tid + s]);
        __syncthreads();
    }
    m = red[0]; __syncthreads();
    float sum = 0.f;
    #pragma unroll
    for (int i = 0; i < 8; ++i) { e[i] = expf(e[i] - m); sum += e[i]; }
    red[tid] = sum; __syncthreads();
    for (int s = 128; s > 0; s >>= 1) {
        if (tid < s) red[tid] += red[tid + s];
        __syncthreads();
    }
    float inv = 1.f / red[0];
    #pragma unroll
    for (int i = 0; i < 8; ++i)
        attn[b * Ln + tid + i * 256] = e[i] * inv;
}

// ------------------- kernel 4: ctx[b,e] = sum_l attn*enc -------------------
__global__ void ctx_kernel(const float* __restrict__ enc,
                           const float* __restrict__ attn,
                           float* __restrict__ ctx) {
    int b = blockIdx.y;
    int e = blockIdx.x * 256 + threadIdx.x;
    const float* ep = enc + (size_t)b * Ln * En + e;
    const float* ap = attn + b * Ln;
    float acc = 0.f;
    for (int l = 0; l < Ln; l += 4) {
        acc += ap[l + 0] * ep[(size_t)(l + 0) * En];
        acc += ap[l + 1] * ep[(size_t)(l + 1) * En];
        acc += ap[l + 2] * ep[(size_t)(l + 2) * En];
        acc += ap[l + 3] * ep[(size_t)(l + 3) * En];
    }
    ctx[b * En + e] = acc;
}

// ---------------------------------------------------------------------------
extern "C" void kernel_launch(void* const* d_in, const int* in_sizes, int n_in,
                              void* d_out, int out_size, void* d_ws, size_t ws_size,
                              hipStream_t stream) {
    const float* dec = (const float*)d_in[0];   // [B,D]
    const float* enc = (const float*)d_in[1];   // [B,L,E]
    const float* Ws  = (const float*)d_in[2];   // [D,D]
    const float* Wh  = (const float*)d_in[3];   // [D,E]
    const float* v   = (const float*)d_in[4];   // [D]

    float* out  = (float*)d_out;
    float* ctx  = out;                  // B*E = 65536 floats
    float* attn = out + Bn * En;        // B*L = 65536 floats

    // workspace layout (needs ~4.6 MB)
    char* ws = (char*)d_ws;
    float*          s_ws   = (float*)ws;                              // 128 KB
    unsigned short* whb    = (unsigned short*)(ws + (128u << 10));    // 4 MB
    float*          scores = (float*)(ws + (128u << 10) + (4u << 20));// 256 KB

    wh_cvt_kernel<<<Dn * En / 4 / 256, 256, 0, stream>>>(Wh, whb);
    sproj_kernel<<<Bn * Dn / 256, 256, 0, stream>>>(dec, Ws, s_ws);

    dim3 g2(Ln / 64, Bn);
    size_t smem2 = (size_t)4 * 16 * En * sizeof(unsigned short)
                 + 256 * sizeof(float);
    score_kernel<<<g2, 128, smem2, stream>>>(enc, whb, s_ws, v, scores);

    softmax_kernel<<<Bn, 256, 0, stream>>>(scores, attn);

    dim3 g4(En / 256, Bn);
    ctx_kernel<<<g4, 256, 0, stream>>>(enc, attn, ctx);
}
